// DeepFeatureExtraction_35064113005001
// MI455X (gfx1250) — matrix operand order repaired and checked by reading, not compiled, not run on hardware
//
#include <hip/hip_runtime.h>

// PointNet++ (SSG, 3 SA + 3 FP + FC) for MI455X / gfx1250, wave32.
// - All einsum/MLP/FC layers run on v_wmma_f32_16x16x32_f16 (f16 in, f32 acc),
//   templated on (K, Cout) so A-fragments live in registers and the MAC chain
//   fully unrolls.
// - Weights are staged once per block into LDS via CDNA5 async global->LDS
//   copies (global_load_async_to_lds_b128 / ASYNCcnt / s_wait_asynccnt), then
//   B-fragments are fed from LDS (ds_load_b128) -- 8 waves share one copy.
// - FPS: one block per batch, LDS tree reduction, dist array lives in L2 (ws).
// - Ball query: wave32 ballot + popcount prefix append (matches top_k(-idx)).
// - BatchNorm (training-mode stats) via LDS float atomics + global atomics.

typedef __attribute__((ext_vector_type(16))) _Float16 v16h;
typedef __attribute__((ext_vector_type(8)))  _Float16 v8h;
typedef __attribute__((ext_vector_type(8)))  float    v8f;

#define NSAMPLE 32

// ---------------------------------------------------------------------------
// (B,3,N) -> (B,N,3)
__global__ void transpose_xyz_kernel(const float* __restrict__ pcl, int B, int N,
                                     float* __restrict__ xyz)
{
    long i = (long)blockIdx.x * blockDim.x + threadIdx.x;
    long total = (long)B * N * 3;
    if (i >= total) return;
    int  c  = (int)(i % 3);
    long bn = i / 3;
    int  n  = (int)(bn % N);
    int  b  = (int)(bn / N);
    xyz[i] = pcl[((long)b * 3 + c) * N + n];
}

// ---------------------------------------------------------------------------
// Farthest point sampling. One block (1024 thr) per batch; dist[] in ws (L2).
__global__ void fps_kernel(const float* __restrict__ xyz, int N, int npoint,
                           float* __restrict__ dist, int* __restrict__ out_idx)
{
    int b = blockIdx.x;
    const float* p = xyz + (long)b * N * 3;
    float* d = dist + (long)b * N;

    __shared__ float rv[32];
    __shared__ int   ri[32];
    __shared__ int   cur;

    for (int i = threadIdx.x; i < N; i += blockDim.x) d[i] = 1e10f;
    if (threadIdx.x == 0) cur = 0;
    __syncthreads();

    int nw   = blockDim.x >> 5;
    int wave = threadIdx.x >> 5;
    int lane = threadIdx.x & 31;

    for (int it = 0; it < npoint; ++it) {
        int far = cur;
        if (threadIdx.x == 0) out_idx[(long)b * npoint + it] = far;
        float cx = p[far * 3 + 0], cy = p[far * 3 + 1], cz = p[far * 3 + 2];

        float best = -1.0f; int besti = 0;
        for (int i = threadIdx.x; i < N; i += blockDim.x) {
            float dx = p[i * 3 + 0] - cx;
            float dy = p[i * 3 + 1] - cy;
            float dz = p[i * 3 + 2] - cz;
            float dd = dx * dx + dy * dy + dz * dz;
            float dm = fminf(d[i], dd);
            d[i] = dm;
            if (dm > best) { best = dm; besti = i; }
        }
        // wave32 reduction
        for (int off = 16; off > 0; off >>= 1) {
            float ov = __shfl_down(best, off);
            int   oi = __shfl_down(besti, off);
            if (ov > best || (ov == best && oi < besti)) { best = ov; besti = oi; }
        }
        if (lane == 0) { rv[wave] = best; ri[wave] = besti; }
        __syncthreads();
        if (threadIdx.x == 0) {
            float bb = rv[0]; int bi = ri[0];
            for (int w = 1; w < nw; ++w)
                if (rv[w] > bb || (rv[w] == bb && ri[w] < bi)) { bb = rv[w]; bi = ri[w]; }
            cur = bi;
        }
        __syncthreads();
    }
}

// ---------------------------------------------------------------------------
__global__ void gather_xyz_kernel(const float* __restrict__ xyz, const int* __restrict__ fidx,
                                  int B, int N, int S, float* __restrict__ nxyz)
{
    long i = (long)blockIdx.x * blockDim.x + threadIdx.x;
    long total = (long)B * S * 3;
    if (i >= total) return;
    int  c  = (int)(i % 3);
    long bs = i / 3;
    int  b  = (int)(bs / S);
    int  j  = fidx[bs];
    nxyz[i] = xyz[((long)b * N + j) * 3 + c];
}

// ---------------------------------------------------------------------------
// One wave per query center: first `ns` in-radius indices (ascending), pad w/ first.
__global__ void ball_query_kernel(const float* __restrict__ xyz, const float* __restrict__ new_xyz,
                                  int B, int N, int S, float r2, int ns, int* __restrict__ idx)
{
    int wid  = (int)(((long)blockIdx.x * blockDim.x + threadIdx.x) >> 5);
    int lane = threadIdx.x & 31;
    if (wid >= B * S) return;
    int b = wid / S;
    const float* p  = xyz + (long)b * N * 3;
    const float* c0 = new_xyz + (long)wid * 3;
    float cx = c0[0], cy = c0[1], cz = c0[2];
    int* o = idx + (long)wid * ns;

    int cnt = 0, first = 0;
    bool haveFirst = false;
    for (int base = 0; base < N && cnt < ns; base += 32) {
        int i = base + lane;
        bool in = false;
        if (i < N) {
            float dx = p[i * 3 + 0] - cx;
            float dy = p[i * 3 + 1] - cy;
            float dz = p[i * 3 + 2] - cz;
            in = (dx * dx + dy * dy + dz * dz <= r2);
        }
        unsigned long long m = __ballot(in);
        if (!haveFirst && m) { first = base + (int)__ffsll((long long)m) - 1; haveFirst = true; }
        int pos = cnt + (int)__popcll(m & ((1ull << lane) - 1ull));
        if (in && pos < ns) o[pos] = i;
        cnt += (int)__popcll(m);
    }
    if (cnt > ns) cnt = ns;
    for (int j = cnt + lane; j < ns; j += 32) o[j] = first;
}

// ---------------------------------------------------------------------------
// Build grouped WMMA input rows: [xyz_j - center (3) | feats_j (D) | zero pad]
__global__ void group_kernel(const float* __restrict__ xyz, const float* __restrict__ feats, int D,
                             const float* __restrict__ new_xyz, const int* __restrict__ idx,
                             int B, int N, int S, int ns, int Kpad, _Float16* __restrict__ out)
{
    long row = (long)blockIdx.x * blockDim.x + threadIdx.x;
    long total = (long)B * S * ns;
    if (row >= total) return;
    long bs = row / ns;
    int  b  = (int)(bs / S);
    int  j  = idx[row];
    const float* p  = xyz + ((long)b * N + j) * 3;
    const float* c0 = new_xyz + bs * 3;
    _Float16* o = out + row * Kpad;
    o[0] = (_Float16)(p[0] - c0[0]);
    o[1] = (_Float16)(p[1] - c0[1]);
    o[2] = (_Float16)(p[2] - c0[2]);
    const float* f = feats + ((long)b * N + j) * D;
    for (int c = 0; c < D; ++c)          o[3 + c] = (_Float16)f[c];
    for (int c = 3 + D; c < Kpad; ++c)   o[c]     = (_Float16)0.0f;
}

// ---------------------------------------------------------------------------
// f32 weight (Cout,Cin) -> f16 (Cout,Kpad) zero-padded
__global__ void convert_w_kernel(const float* __restrict__ W, int Cout, int Cin, int Kpad,
                                 _Float16* __restrict__ out)
{
    int i = blockIdx.x * blockDim.x + threadIdx.x;
    if (i >= Cout * Kpad) return;
    int k = i % Kpad, o = i / Kpad;
    out[i] = (k < Cin) ? (_Float16)W[o * Cin + k] : (_Float16)0.0f;
}

// ---------------------------------------------------------------------------
// Y[M,COUT] = X[M,K] * W[COUT,K]^T + bias   (f16 operands, f32 accumulate)
// - W (COUT*K*2 bytes, <= 16 KB) staged block-wide into LDS with CDNA5 async
//   global->LDS b128 copies; B-fragments then come from LDS (ds_load_b128).
// - A-fragments (16-bit 16x32 layout) loaded once per wave and reused across
//   all COUT/16 column tiles.
__attribute__((noinline))
__device__ void async_copy16_to_lds(unsigned ldsAddr, const char* gaddr)
{
    asm volatile("global_load_async_to_lds_b128 %0, %1, off"
                 :: "v"(ldsAddr), "v"(gaddr) : "memory");
}

template <int K, int COUT>
__global__ void wmma_gemm_kernel(const _Float16* __restrict__ X,
                                 const _Float16* __restrict__ W,
                                 const float* __restrict__ bias,
                                 float* __restrict__ Y,
                                 int M)
{
    __shared__ __align__(16) _Float16 ldsW[COUT * K];

    // ---- async stage the full weight matrix into LDS (shared by 8 waves) ----
    {
        const int totalBytes = COUT * K * 2;
        const unsigned ldsBase = (unsigned)(uintptr_t)(&ldsW[0]);
        const char* gsrc = (const char*)W;
        for (int off = (int)threadIdx.x * 16; off < totalBytes; off += (int)blockDim.x * 16)
            async_copy16_to_lds(ldsBase + (unsigned)off, gsrc + off);
        asm volatile("s_wait_asynccnt 0" ::: "memory");
        __syncthreads();
    }

    const int lane = threadIdx.x & 31;
    const int wave = threadIdx.x >> 5;
    const long m0 = ((long)blockIdx.x * 8 + wave) * 16;   // 8 waves / block
    if (m0 >= M) return;                                  // wave-uniform

    const int kbaseA = (lane >> 4) * 8;     // A frag: 16-bit 16x32 layout
    const int kbaseB = (lane >> 4) * 8;     // B frag: 16-bit 32x16 layout (same K order as A)
    const int ncol   = lane & 15;

    union Frag { v16h v; v8h h[2]; };

    // ---- load all A fragments once (K/32 <= 4 fragments, stays in VGPRs) ----
    constexpr int NK = K / 32;
    Frag a[NK];
    {
        const _Float16* xbase = X + (m0 + (lane & 15)) * (long)K;
#pragma unroll
        for (int kk = 0; kk < NK; ++kk) {
            a[kk].h[0] = *(const v8h*)(xbase + kk * 32 + kbaseA);
            a[kk].h[1] = *(const v8h*)(xbase + kk * 32 + kbaseA + 16);
        }
    }

    // ---- column tiles of 16 over COUT, B fragments from LDS ----
#pragma unroll
    for (int ct = 0; ct < COUT / 16; ++ct) {
        const _Float16* wb = &ldsW[(ct * 16 + ncol) * K];
        v8f acc = {};
#pragma unroll
        for (int kk = 0; kk < NK; ++kk) {
            Frag bfr;
            bfr.h[0] = *(const v8h*)(wb + kk * 32 + kbaseB);
            bfr.h[1] = *(const v8h*)(wb + kk * 32 + kbaseB + 16);
            acc = __builtin_amdgcn_wmma_f32_16x16x32_f16(
                false, a[kk].v, false, bfr.v, (short)0, acc, false, false);
        }
        // C/D layout: VGPR i -> row m0 + (lane>>4)*8 + i, col = ct*16 + (lane&15)
        int n  = ct * 16 + ncol;
        float bv = bias ? bias[n] : 0.0f;
        long rbase = m0 + (long)((lane >> 4) * 8);
#pragma unroll
        for (int i = 0; i < 8; ++i)
            Y[(rbase + i) * COUT + n] = acc[i] + bv;
    }
}

// ---------------------------------------------------------------------------
__global__ void zero_kernel(float* __restrict__ p, int n)
{
    int i = blockIdx.x * blockDim.x + threadIdx.x;
    if (i < n) p[i] = 0.0f;
}

// Per-channel sum / sumsq via LDS float atomics, then one global atomic per ch.
__global__ void bn_stats_kernel(const float* __restrict__ Y, long M, int C,
                                float* __restrict__ sums, float* __restrict__ sumsq)
{
    __shared__ float ls[64], lq[64];
    for (int c = threadIdx.x; c < C; c += blockDim.x) { ls[c] = 0.0f; lq[c] = 0.0f; }
    __syncthreads();
    long total = M * C;
    for (long i = (long)blockIdx.x * blockDim.x + threadIdx.x; i < total;
         i += (long)gridDim.x * blockDim.x) {
        float v = Y[i];
        int c = (int)(i % C);
        atomicAdd(&ls[c], v);
        atomicAdd(&lq[c], v * v);
    }
    __syncthreads();
    for (int c = threadIdx.x; c < C; c += blockDim.x) {
        atomicAdd(&sums[c],  ls[c]);
        atomicAdd(&sumsq[c], lq[c]);
    }
}

// Normalize + affine + ReLU; write f16 (next GEMM input) and optional f32.
__global__ void bn_apply_kernel(const float* __restrict__ Y,
                                const float* __restrict__ sums, const float* __restrict__ sumsq,
                                const float* __restrict__ gamma, const float* __restrict__ beta,
                                long M, int C,
                                _Float16* __restrict__ outH, float* __restrict__ outF)
{
    long i = (long)blockIdx.x * blockDim.x + threadIdx.x;
    long total = M * C;
    if (i >= total) return;
    int c = (int)(i % C);
    float mean = sums[c] / (float)M;
    float var  = sumsq[c] / (float)M - mean * mean;
    float v = (Y[i] - mean) * rsqrtf(var + 1e-5f) * gamma[c] + beta[c];
    v = fmaxf(v, 0.0f);
    outH[i] = (_Float16)v;
    if (outF) outF[i] = v;
}

// ---------------------------------------------------------------------------
__global__ void maxpool_kernel(const float* __restrict__ X, int BS, int ns, int C,
                               float* __restrict__ out)
{
    long i = (long)blockIdx.x * blockDim.x + threadIdx.x;
    if (i >= (long)BS * C) return;
    int  c  = (int)(i % C);
    long bs = i / C;
    const float* p = X + (bs * ns) * (long)C + c;
    float m = p[0];
    for (int k = 1; k < ns; ++k) m = fmaxf(m, p[(long)k * C]);
    out[i] = m;
}

// ---------------------------------------------------------------------------
// 3-NN inverse-distance interpolation + concat -> f16 GEMM input rows.
__global__ void fp_interp_kernel(const float* __restrict__ xyz1, const float* __restrict__ xyz2,
                                 const float* __restrict__ feats1, int D1,
                                 const float* __restrict__ feats2, int D2,
                                 int B, int N1, int N2, int Kpad, _Float16* __restrict__ out)
{
    long i = (long)blockIdx.x * blockDim.x + threadIdx.x;
    if (i >= (long)B * N1) return;
    int b = (int)(i / N1);
    float px = xyz1[i * 3 + 0], py = xyz1[i * 3 + 1], pz = xyz1[i * 3 + 2];
    const float* q = xyz2 + (long)b * N2 * 3;
    float d0 = 1e30f, d1 = 1e30f, d2 = 1e30f;
    int   i0 = 0, i1 = 0, i2 = 0;
    for (int j = 0; j < N2; ++j) {
        float dx = q[j * 3 + 0] - px;
        float dy = q[j * 3 + 1] - py;
        float dz = q[j * 3 + 2] - pz;
        float d = dx * dx + dy * dy + dz * dz;
        if (d < d0)      { d2 = d1; i2 = i1; d1 = d0; i1 = i0; d0 = d; i0 = j; }
        else if (d < d1) { d2 = d1; i2 = i1; d1 = d;  i1 = j; }
        else if (d < d2) { d2 = d;  i2 = j; }
    }
    float w0 = 1.0f / (d0 + 1e-8f);
    float w1 = 1.0f / (d1 + 1e-8f);
    float w2 = 1.0f / (d2 + 1e-8f);
    float wsum = w0 + w1 + w2;
    w0 /= wsum; w1 /= wsum; w2 /= wsum;

    _Float16* o = out + i * Kpad;
    for (int c = 0; c < D1; ++c) o[c] = (_Float16)feats1[i * (long)D1 + c];
    const float* f2 = feats2 + (long)b * N2 * D2;
    for (int c = 0; c < D2; ++c)
        o[D1 + c] = (_Float16)(w0 * f2[(long)i0 * D2 + c] +
                               w1 * f2[(long)i1 * D2 + c] +
                               w2 * f2[(long)i2 * D2 + c]);
    for (int c = D1 + D2; c < Kpad; ++c) o[c] = (_Float16)0.0f;
}

// ===========================================================================
// Host orchestration
// ===========================================================================
namespace {

struct Bump {
    char* base; size_t off;
    void* take(size_t bytes) {
        size_t o = (off + 255) & ~(size_t)255;
        off = o + bytes;
        return base + o;
    }
};

struct Layer {
    const float *W, *b, *g, *bt;
    int Cin, Cout, Kpad;
    _Float16* Wh;
};

inline int divUp(long a, int b) { return (int)((a + b - 1) / b); }

void launch_gemm(const _Float16* X, const _Float16* Wh, const float* bias, float* Y,
                 long M, int K, int Cout, hipStream_t s)
{
    int blocks = (int)(M / 128);
#define GEMM_CASE(KK, CC)                                                         \
    if (K == KK && Cout == CC) {                                                  \
        wmma_gemm_kernel<KK, CC><<<blocks, 256, 0, s>>>(X, Wh, bias, Y, (int)M);  \
        return;                                                                   \
    }
    GEMM_CASE(32, 32)
    GEMM_CASE(32, 64)
    GEMM_CASE(64, 32)
    GEMM_CASE(64, 64)
    GEMM_CASE(96, 32)
    GEMM_CASE(96, 64)
    GEMM_CASE(128, 64)
#undef GEMM_CASE
}

void run_mlp(const Layer* Ls, int nl, const _Float16* Xin, long M,
             _Float16* Hbuf, float* preact, float* stats, float* outF_last, hipStream_t s)
{
    for (int l = 0; l < nl; ++l) {
        const Layer& L = Ls[l];
        const _Float16* in = (l == 0) ? Xin : Hbuf;
        launch_gemm(in, L.Wh, L.b, preact, M, L.Kpad, L.Cout, s);
        zero_kernel<<<1, 128, 0, s>>>(stats, 128);
        bn_stats_kernel<<<512, 256, 0, s>>>(preact, M, L.Cout, stats, stats + 64);
        long total = M * L.Cout;
        bn_apply_kernel<<<divUp(total, 256), 256, 0, s>>>(
            preact, stats, stats + 64, L.g, L.bt, M, L.Cout,
            Hbuf, (l == nl - 1) ? outF_last : (float*)nullptr);
    }
}

void run_sa(const float* xyz_in, int N, const float* feats_in, int D,
            int S, float radius, const Layer* Ls,
            int* fidx, float* dist, int* bidx, float* xyz_out,
            _Float16* gemmIn, _Float16* Hbuf, float* preact, float* stats,
            float* actF, float* f_out, hipStream_t s)
{
    const int B = 2;
    fps_kernel<<<B, 1024, 0, s>>>(xyz_in, N, S, dist, fidx);
    gather_xyz_kernel<<<divUp((long)B * S * 3, 256), 256, 0, s>>>(xyz_in, fidx, B, N, S, xyz_out);
    ball_query_kernel<<<divUp((long)B * S * 32, 256), 256, 0, s>>>(
        xyz_in, xyz_out, B, N, S, radius * radius, NSAMPLE, bidx);
    long M = (long)B * S * NSAMPLE;
    group_kernel<<<divUp(M, 256), 256, 0, s>>>(
        xyz_in, feats_in, D, xyz_out, bidx, B, N, S, NSAMPLE, Ls[0].Kpad, gemmIn);
    run_mlp(Ls, 2, gemmIn, M, Hbuf, preact, stats, actF, s);
    int C = Ls[1].Cout;
    maxpool_kernel<<<divUp((long)B * S * C, 256), 256, 0, s>>>(actF, B * S, NSAMPLE, C, f_out);
}

void run_fp(const float* xyz1, int N1, const float* xyz2, int N2,
            const float* feats1, int D1, const float* feats2, int D2,
            const Layer* Ls, int nl,
            _Float16* gemmIn, _Float16* Hbuf, float* preact, float* stats,
            float* f_out, hipStream_t s)
{
    const int B = 2;
    long Mpts = (long)B * N1;
    fp_interp_kernel<<<divUp(Mpts, 256), 256, 0, s>>>(
        xyz1, xyz2, feats1, D1, feats2, D2, B, N1, N2, Ls[0].Kpad, gemmIn);
    run_mlp(Ls, nl, gemmIn, Mpts, Hbuf, preact, stats, f_out, s);
}

} // namespace

extern "C" void kernel_launch(void* const* d_in, const int* in_sizes, int n_in,
                              void* d_out, int out_size, void* d_ws, size_t ws_size,
                              hipStream_t stream)
{
    (void)in_sizes; (void)n_in; (void)out_size; (void)ws_size;

    const int B = 2, N0 = 16384, S1 = 4096, S2 = 1024, S3 = 512;
    const float* pcl = (const float*)d_in[0];

    int t = 1;
    auto grab4 = [&](int Cin, int Cout) {
        Layer L;
        L.W  = (const float*)d_in[t + 0];
        L.b  = (const float*)d_in[t + 1];
        L.g  = (const float*)d_in[t + 2];
        L.bt = (const float*)d_in[t + 3];
        t += 4;
        L.Cin = Cin; L.Cout = Cout;
        L.Kpad = ((Cin + 31) / 32) * 32;
        L.Wh = nullptr;
        return L;
    };
    Layer sa1[2] = { grab4(6, 32),   grab4(32, 32) };
    Layer sa2[2] = { grab4(35, 32),  grab4(32, 64) };
    Layer sa3[2] = { grab4(67, 64),  grab4(64, 64) };
    Layer fp3[2] = { grab4(128, 64), grab4(64, 64) };
    Layer fp2[2] = { grab4(96, 32),  grab4(32, 32) };
    Layer fp1[3] = { grab4(32, 32),  grab4(32, 32), grab4(32, 32) };
    Layer fc;
    fc.W = (const float*)d_in[t + 0];
    fc.b = (const float*)d_in[t + 1];
    fc.g = nullptr; fc.bt = nullptr;
    fc.Cin = 32; fc.Cout = 32; fc.Kpad = 32; fc.Wh = nullptr;

    Bump ws{ (char*)d_ws, 0 };

    float* xyz0  = (float*)ws.take((size_t)B * N0 * 3 * 4);
    float* dist  = (float*)ws.take((size_t)B * N0 * 4);
    int*   fidx1 = (int*)  ws.take((size_t)B * S1 * 4);
    int*   fidx2 = (int*)  ws.take((size_t)B * S2 * 4);
    int*   fidx3 = (int*)  ws.take((size_t)B * S3 * 4);
    float* xyz1  = (float*)ws.take((size_t)B * S1 * 3 * 4);
    float* xyz2  = (float*)ws.take((size_t)B * S2 * 3 * 4);
    float* xyz3  = (float*)ws.take((size_t)B * S3 * 3 * 4);
    int*   bidx  = (int*)  ws.take((size_t)B * S1 * NSAMPLE * 4);

    const long Mmax = (long)B * S1 * NSAMPLE;                 // 262144 rows
    _Float16* gemmIn = (_Float16*)ws.take((size_t)Mmax * 64 * 2);
    _Float16* Hbuf   = (_Float16*)ws.take((size_t)Mmax * 64 * 2);
    float*    preact = (float*)   ws.take((size_t)Mmax * 32 * 4);
    float*    actF   = (float*)   ws.take((size_t)Mmax * 32 * 4);
    float*    stats  = (float*)   ws.take(128 * 4);

    float* f_s1  = (float*)ws.take((size_t)B * S1 * 32 * 4);
    float* f_s2  = (float*)ws.take((size_t)B * S2 * 64 * 4);
    float* f_s3  = (float*)ws.take((size_t)B * S3 * 64 * 4);
    float* f_s2b = (float*)ws.take((size_t)B * S2 * 64 * 4);
    float* f_s1b = (float*)ws.take((size_t)B * S1 * 32 * 4);

    // f16 weight staging (zero-padded to Kpad)
    auto convW = [&](Layer& L) {
        L.Wh = (_Float16*)ws.take((size_t)L.Cout * L.Kpad * 2);
        int tot = L.Cout * L.Kpad;
        convert_w_kernel<<<divUp(tot, 256), 256, 0, stream>>>(L.W, L.Cout, L.Cin, L.Kpad, L.Wh);
    };
    for (int i = 0; i < 2; ++i) { convW(sa1[i]); convW(sa2[i]); convW(sa3[i]); }
    for (int i = 0; i < 2; ++i) { convW(fp3[i]); convW(fp2[i]); }
    for (int i = 0; i < 3; ++i) { convW(fp1[i]); }
    convW(fc);

    // (B,3,N) -> (B,N,3); stage-0 points == xyz
    transpose_xyz_kernel<<<divUp((long)B * N0 * 3, 256), 256, 0, stream>>>(pcl, B, N0, xyz0);

    // Set abstraction stages
    run_sa(xyz0, N0, xyz0, 3,  S1, 0.1f, sa1, fidx1, dist, bidx, xyz1,
           gemmIn, Hbuf, preact, stats, actF, f_s1, stream);
    run_sa(xyz1, S1, f_s1, 32, S2, 0.2f, sa2, fidx2, dist, bidx, xyz2,
           gemmIn, Hbuf, preact, stats, actF, f_s2, stream);
    run_sa(xyz2, S2, f_s2, 64, S3, 0.4f, sa3, fidx3, dist, bidx, xyz3,
           gemmIn, Hbuf, preact, stats, actF, f_s3, stream);

    // Feature propagation stages
    run_fp(xyz2, S2, xyz3, S3, f_s2, 64, f_s3, 64, fp3, 2,
           gemmIn, Hbuf, preact, stats, f_s2b, stream);
    run_fp(xyz1, S1, xyz2, S2, f_s1, 32, f_s2b, 64, fp2, 2,
           gemmIn, Hbuf, preact, stats, f_s1b, stream);
    run_fp(xyz0, N0, xyz1, S1, nullptr, 0, f_s1b, 32, fp1, 3,
           gemmIn, Hbuf, preact, stats, nullptr, stream);

    // Final FC: (B*N0, 32) @ fc_w^T + fc_b -> d_out  (Hbuf holds fp1 output f16)
    launch_gemm(Hbuf, fc.Wh, fc.b, (float*)d_out, (long)B * N0, 32, 32, stream);
}
